// Block_6219112644962
// MI455X (gfx1250) — matrix operand order repaired and checked by reading, not compiled, not run on hardware
//
#include <hip/hip_runtime.h>
#include <hip/hip_bf16.h>

typedef __attribute__((ext_vector_type(16))) _Float16 v16h;
typedef __attribute__((ext_vector_type(8)))  _Float16 v8h;
typedef __attribute__((ext_vector_type(8)))  float    v8f;

#if defined(__gfx1250__) && __has_builtin(__builtin_amdgcn_tensor_load_to_lds) && \
    __has_builtin(__builtin_amdgcn_s_wait_tensorcnt)
#define USE_TDM 1
#else
#define USE_TDM 0
#endif

#if USE_TDM
typedef __attribute__((ext_vector_type(4))) unsigned int tdm_u32x4;
typedef __attribute__((ext_vector_type(8))) int          tdm_i32x8;
typedef __attribute__((ext_vector_type(4))) int          tdm_i32x4;

// Issue one 2D TDM tile load: tile_x (contiguous, elems) x tile_y rows of f16,
// row stride = stride_elems, into LDS at lds_off with HW padding
// (pad_interval: 0=2dw,1=4dw,2=8dw,3=16dw,4=32dw ; pad_amount: dwords-1).
__device__ __forceinline__ void tdm_load_2d(const _Float16* g, unsigned lds_off,
                                            int stride_elems, int tile_x, int tile_y,
                                            int pad_interval, int pad_amount) {
  unsigned long long ga = (unsigned long long)(size_t)g;
  tdm_u32x4 g0;
  g0[0] = 1u;                                              // count=1, user mode
  g0[1] = lds_off;                                         // lds_addr (bytes)
  g0[2] = (unsigned)ga;                                    // global_addr[31:0]
  g0[3] = (unsigned)((ga >> 32) & 0x01ffffffu) | (2u << 30); // addr[56:32] | type=2
  tdm_i32x8 g1;
  g1[0] = (1 << 16) | (1 << 20) | (pad_interval << 22) | (pad_amount << 25); // 2B elems, pad_enable
  g1[1] = (stride_elems & 0xffff) << 16;                   // tensor_dim0 lo16
  g1[2] = ((stride_elems >> 16) & 0xffff) | ((tile_y & 0xffff) << 16); // dim0 hi | dim1 lo
  g1[3] = (tile_x & 0xffff) << 16;                         // dim1 hi=0 | tile_dim0
  g1[4] = (tile_y & 0xffff);                               // tile_dim1 | tile_dim2=0
  g1[5] = stride_elems;                                    // tensor_dim0_stride lo32
  g1[6] = 0;                                               // stride hi | dim1_stride lo
  g1[7] = 0;
  tdm_i32x4 z4 = {0, 0, 0, 0};
  tdm_i32x8 z8 = {0, 0, 0, 0, 0, 0, 0, 0};
  __builtin_amdgcn_tensor_load_to_lds(g0, g1, z4, z4, z8, 0);
}
#endif

// ---------------------------------------------------------------------------
// WMMA fragment loaders per CDNA5 ISA layouts (wave32).
// ---------------------------------------------------------------------------
__device__ __forceinline__ v16h frag_a_16x32(const _Float16* __restrict__ p, int ld) {
  const int lane = threadIdx.x & 31;
  const _Float16* r = p + (size_t)(lane & 15) * ld + 8 * (lane >> 4);
  v16h a;
#pragma unroll
  for (int j = 0; j < 8; ++j) a[j] = r[j];
#pragma unroll
  for (int j = 0; j < 8; ++j) a[8 + j] = r[16 + j];
  return a;
}

__device__ __forceinline__ v16h frag_b_32x16(const _Float16* __restrict__ p, int ld) {
  const int lane = threadIdx.x & 31;
  const _Float16* r = p + (size_t)(lane & 15) * ld + 8 * (lane >> 4);
  v16h b;
#pragma unroll
  for (int j = 0; j < 8; ++j) b[j] = r[j];
#pragma unroll
  for (int j = 0; j < 8; ++j) b[8 + j] = r[16 + j];
  return b;
}

__device__ __forceinline__ v8f wmma_f16(v16h a, v16h b, v8f c) {
  return __builtin_amdgcn_wmma_f32_16x16x32_f16(false, a, false, b, (short)0, c, false, false);
}

// ---------------------------------------------------------------------------
// GEMM: O[m,n] = sum_k X[m,k] * W[n,k]  (X: MxK f16, W: NxK f16)
// mode 0: outH = f16(O) ; mode 1: outH = f16(gelu(O+bias)) ; mode 2: outF = O+bias+resid
// Tile 128x128x32, 256 threads = 8 waves (2x4), 64x32 per wave, TDM-staged LDS.
// ---------------------------------------------------------------------------
#define TM 128
#define TN 128
#define TK 32
#define LDT (TK + 8)

__global__ __launch_bounds__(256, 2) void gemm_wmma_kernel(
    const _Float16* __restrict__ X, const _Float16* __restrict__ W,
    _Float16* __restrict__ outH, float* __restrict__ outF,
    const float* __restrict__ bias, const float* __restrict__ resid,
    int M, int N, int K, int mode) {
  const int tid  = threadIdx.x;
  const int wave = tid >> 5;
  const int lane = tid & 31;
  const int wm = wave >> 2, wn = wave & 3;
  const int m0 = blockIdx.y * TM;
  const int n0 = blockIdx.x * TN;

  __shared__ _Float16 sA[2][TM * LDT];
  __shared__ _Float16 sB[2][TN * LDT];

#if USE_TDM
  auto issue_stage = [&](int buf, int k0) {
    if (wave == 0) {
      tdm_load_2d(&X[(size_t)m0 * K + k0], (unsigned)(size_t)&sA[buf][0], K, TK, TM, 3, 3);
      tdm_load_2d(&W[(size_t)n0 * K + k0], (unsigned)(size_t)&sB[buf][0], K, TK, TN, 3, 3);
    }
  };
  auto wait_stage = [&]() {
    if (wave == 0) __builtin_amdgcn_s_wait_tensorcnt(0);
  };
#else
  auto issue_stage = [&](int buf, int k0) {
#pragma unroll
    for (int u = 0; u < (TM * TK / 8) / 256; ++u) {
      const int i = tid + u * 256;
      const int r = i >> 2, c = (i & 3) << 3;
      const _Float16* g = &X[(size_t)(m0 + r) * K + k0 + c];
      *(uint4*)&sA[buf][r * LDT + c] = *(const uint4*)g;
      if (k0 + 2 * TK < K) __builtin_prefetch(g + 2 * TK, 0, 0);
    }
#pragma unroll
    for (int u = 0; u < (TN * TK / 8) / 256; ++u) {
      const int i = tid + u * 256;
      const int r = i >> 2, c = (i & 3) << 3;
      const _Float16* g = &W[(size_t)(n0 + r) * K + k0 + c];
      *(uint4*)&sB[buf][r * LDT + c] = *(const uint4*)g;
      if (k0 + 2 * TK < K) __builtin_prefetch(g + 2 * TK, 0, 0);
    }
  };
  auto wait_stage = [&]() {};
#endif

  v8f acc[4][2] = {};
  issue_stage(0, 0);
  wait_stage();
  __syncthreads();

  const int nk = K / TK;
  for (int kt = 0; kt < nk; ++kt) {
    const int cur = kt & 1;
    if (kt + 1 < nk) issue_stage(cur ^ 1, (kt + 1) * TK);

    v16h a[4], bf[2];
#pragma unroll
    for (int i = 0; i < 4; ++i)
      a[i] = frag_a_16x32(&sA[cur][(wm * 64 + i * 16) * LDT], LDT);
#pragma unroll
    for (int j = 0; j < 2; ++j)
      bf[j] = frag_b_32x16(&sB[cur][(wn * 32 + j * 16) * LDT], LDT);
#pragma unroll
    for (int i = 0; i < 4; ++i)
#pragma unroll
      for (int j = 0; j < 2; ++j)
        acc[i][j] = wmma_f16(a[i], bf[j], acc[i][j]);

    if (kt + 1 < nk) wait_stage();
    __syncthreads();
  }

  const int hi = lane >> 4, ln = lane & 15;
#pragma unroll
  for (int i = 0; i < 4; ++i)
#pragma unroll
    for (int j = 0; j < 2; ++j)
#pragma unroll
      for (int r = 0; r < 8; ++r) {
        const int row = m0 + wm * 64 + i * 16 + hi * 8 + r;
        const int col = n0 + wn * 32 + j * 16 + ln;
        float v = acc[i][j][r];
        if (mode == 1) {
          v += bias[col];
          v = 0.5f * v * (1.0f + erff(v * 0.70710678118654752f));
          outH[(size_t)row * N + col] = (_Float16)v;
        } else if (mode == 2) {
          v += bias[col] + resid[(size_t)row * N + col];
          outF[(size_t)row * N + col] = v;
        } else {
          outH[(size_t)row * N + col] = (_Float16)v;
        }
      }
}

// ---------------------------------------------------------------------------
// Flash attention for one (b, h, 128-row q tile). 8 waves, 16 q rows each.
// ---------------------------------------------------------------------------
#define HD  64
#define KT  128
#define LDK (HD + 8)
#define LDV (KT + 8)
#define LDP (KT + 8)

__global__ __launch_bounds__(256, 1) void attn_flash_kernel(
    const _Float16* __restrict__ qkv, _Float16* __restrict__ out,
    int N, int C, int H) {
  const int bh = blockIdx.x;
  const int b = bh / H, h = bh % H;
  const int q0 = blockIdx.y * 128;
  const int tid = threadIdx.x, wave = tid >> 5, lane = tid & 31;
  const int hi = lane >> 4, ln = lane & 15;
  const int C3 = 3 * C;

  const _Float16* Qg = qkv + (size_t)b * N * C3 + 0 * C + h * HD;
  const _Float16* Kg = qkv + (size_t)b * N * C3 + 1 * C + h * HD;
  const _Float16* Vg = qkv + (size_t)b * N * C3 + 2 * C + h * HD;

  __shared__ _Float16 sK[KT * LDK];      // [key][d]
  __shared__ _Float16 sV[HD * LDV];      // transposed: [d][key]
  __shared__ _Float16 sP[8][16 * LDP];   // per-wave P tile [row][key]

  v16h qf[2];
#pragma unroll
  for (int kk = 0; kk < 2; ++kk)
    qf[kk] = frag_a_16x32(Qg + (size_t)(q0 + wave * 16) * C3 + kk * 32, C3);

  v8f o[4] = {};
  float mrow[8], lrow[8];
#pragma unroll
  for (int r = 0; r < 8; ++r) { mrow[r] = -__builtin_huge_valf(); lrow[r] = 0.f; }
  const float scale = 0.125f;  // hd^-0.5

  for (int kt0 = 0; kt0 < N; kt0 += KT) {
    __syncthreads();  // previous tile fully consumed
#if USE_TDM
    if (wave == 0)   // K tile: 64 halves/row = 32 dwords -> pad_interval=4, +4 dwords
      tdm_load_2d(&Kg[(size_t)kt0 * C3], (unsigned)(size_t)&sK[0], C3, HD, KT, 4, 3);
#else
#pragma unroll
    for (int u = 0; u < (KT * HD / 8) / 256; ++u) {
      const int i = tid + u * 256;
      const int r = i >> 3, c = (i & 7) << 3;
      *(uint4*)&sK[r * LDK + c] = *(const uint4*)&Kg[(size_t)(kt0 + r) * C3 + c];
    }
#endif
    // stage V tile transposed -> [d][key] (manual: TDM cannot transpose)
#pragma unroll
    for (int u = 0; u < (KT * HD / 8) / 256; ++u) {
      const int i = tid + u * 256;
      const int r = i >> 3, c = (i & 7) << 3;
      v8h vv = *(const v8h*)&Vg[(size_t)(kt0 + r) * C3 + c];
#pragma unroll
      for (int t = 0; t < 8; ++t) sV[(c + t) * LDV + r] = vv[t];
    }
#if USE_TDM
    if (wave == 0) __builtin_amdgcn_s_wait_tensorcnt(0);
#endif
    __syncthreads();

    // S = Q K^T (16 x 128 per wave)
    v8f sacc[8] = {};
#pragma unroll
    for (int j = 0; j < 8; ++j)
#pragma unroll
      for (int kk = 0; kk < 2; ++kk) {
        v16h bk = frag_b_32x16(&sK[(j * 16) * LDK + kk * 32], LDK);
        sacc[j] = wmma_f16(qf[kk], bk, sacc[j]);
      }

    // online softmax; row r_local = hi*8 + r lives across its 16-lane group
#pragma unroll
    for (int r = 0; r < 8; ++r) {
      float mx = -__builtin_huge_valf();
#pragma unroll
      for (int j = 0; j < 8; ++j) mx = fmaxf(mx, sacc[j][r]);
      mx *= scale;
#pragma unroll
      for (int msk = 1; msk < 16; msk <<= 1) mx = fmaxf(mx, __shfl_xor(mx, msk));
      const float mnew = fmaxf(mrow[r], mx);
      const float corr = __expf(mrow[r] - mnew);
      mrow[r] = mnew;
      float rs = 0.f;
#pragma unroll
      for (int j = 0; j < 8; ++j) {
        float p = __expf(sacc[j][r] * scale - mnew);
        sacc[j][r] = p;
        rs += p;
      }
#pragma unroll
      for (int msk = 1; msk < 16; msk <<= 1) rs += __shfl_xor(rs, msk);
      lrow[r] = lrow[r] * corr + rs;
#pragma unroll
      for (int jd = 0; jd < 4; ++jd) o[jd][r] *= corr;
    }

    // re-fragment P through LDS (C-layout -> A-layout)
    _Float16* pw = sP[wave];
#pragma unroll
    for (int j = 0; j < 8; ++j)
#pragma unroll
      for (int r = 0; r < 8; ++r)
        pw[(hi * 8 + r) * LDP + j * 16 + ln] = (_Float16)sacc[j][r];
    __syncthreads();

    // O += P @ V   (K dim = 128 keys, 4 wmma k-steps)
#pragma unroll
    for (int jd = 0; jd < 4; ++jd)
#pragma unroll
      for (int kk = 0; kk < 4; ++kk) {
        v16h pa = frag_a_16x32(pw + kk * 32, LDP);
        v16h vb = frag_b_32x16(&sV[(jd * 16) * LDV + kk * 32], LDV);
        o[jd] = wmma_f16(pa, vb, o[jd]);
      }
  }

#pragma unroll
  for (int jd = 0; jd < 4; ++jd)
#pragma unroll
    for (int r = 0; r < 8; ++r) {
      const int row = q0 + wave * 16 + hi * 8 + r;
      const int col = h * HD + jd * 16 + ln;
      out[((size_t)b * N + row) * C + col] = (_Float16)(o[jd][r] / lrow[r]);
    }
}

// ---------------------------------------------------------------------------
// LayerNorm over C=1024, one block (256 thr) per row, f16 output.
// ---------------------------------------------------------------------------
__global__ __launch_bounds__(256) void ln_f16_kernel(
    const float* __restrict__ x, const float* __restrict__ g,
    const float* __restrict__ b, _Float16* __restrict__ y, int C) {
  const int row = blockIdx.x;
  const float* xr = x + (size_t)row * C;
  float v[4], s = 0.f, s2 = 0.f;
#pragma unroll
  for (int i = 0; i < 4; ++i) {
    v[i] = xr[threadIdx.x + i * 256];
    s += v[i]; s2 += v[i] * v[i];
  }
#pragma unroll
  for (int m = 16; m >= 1; m >>= 1) { s += __shfl_xor(s, m); s2 += __shfl_xor(s2, m); }
  __shared__ float w1[8], w2[8];
  if ((threadIdx.x & 31) == 0) { w1[threadIdx.x >> 5] = s; w2[threadIdx.x >> 5] = s2; }
  __syncthreads();
  s = w1[threadIdx.x & 7]; s2 = w2[threadIdx.x & 7];
#pragma unroll
  for (int m = 4; m >= 1; m >>= 1) { s += __shfl_xor(s, m); s2 += __shfl_xor(s2, m); }
  const float mu = s / C;
  const float rstd = rsqrtf(s2 / C - mu * mu + 1e-5f);
#pragma unroll
  for (int i = 0; i < 4; ++i) {
    const int c = threadIdx.x + i * 256;
    y[(size_t)row * C + c] = (_Float16)((v[i] - mu) * rstd * g[c] + b[c]);
  }
}

// f32 -> f16 cast (vector-4)
__global__ __launch_bounds__(256) void cast_f32_f16_kernel(
    const float* __restrict__ src, _Float16* __restrict__ dst, int n4) {
  const int i = blockIdx.x * 256 + threadIdx.x;
  if (i < n4) {
    float4 f = ((const float4*)src)[i];
    union { _Float16 h[4]; uint2 u; } o;
    o.h[0] = (_Float16)f.x; o.h[1] = (_Float16)f.y;
    o.h[2] = (_Float16)f.z; o.h[3] = (_Float16)f.w;
    ((uint2*)dst)[i] = o.u;
  }
}

// ---------------------------------------------------------------------------
extern "C" void kernel_launch(void* const* d_in, const int* in_sizes, int n_in,
                              void* d_out, int out_size, void* d_ws, size_t ws_size,
                              hipStream_t stream) {
  (void)in_sizes; (void)n_in; (void)out_size; (void)ws_size;
  const float* x      = (const float*)d_in[0];
  const float* w_qkv  = (const float*)d_in[1];
  const float* w_proj = (const float*)d_in[2];
  const float* b_proj = (const float*)d_in[3];
  const float* ln1_g  = (const float*)d_in[4];
  const float* ln1_b  = (const float*)d_in[5];
  const float* ln2_g  = (const float*)d_in[6];
  const float* ln2_b  = (const float*)d_in[7];
  const float* w_fc1  = (const float*)d_in[8];
  const float* b_fc1  = (const float*)d_in[9];
  const float* w_fc2  = (const float*)d_in[10];
  const float* b_fc2  = (const float*)d_in[11];
  float* out = (float*)d_out;

  const int B = 2, N = 2048, C = 1024, H = 16, HID = 4096;
  const int M = B * N;  // 4096

  char* ws = (char*)d_ws;
  _Float16* wqkv_h  = (_Float16*)ws;  ws += (size_t)3 * C * C * sizeof(_Float16);
  _Float16* wproj_h = (_Float16*)ws;  ws += (size_t)C * C * sizeof(_Float16);
  _Float16* wfc1_h  = (_Float16*)ws;  ws += (size_t)HID * C * sizeof(_Float16);
  _Float16* wfc2_h  = (_Float16*)ws;  ws += (size_t)C * HID * sizeof(_Float16);
  _Float16* ln_h    = (_Float16*)ws;  ws += (size_t)M * C * sizeof(_Float16);
  _Float16* qkv_h   = (_Float16*)ws;  ws += (size_t)M * 3 * C * sizeof(_Float16);
  _Float16* attn_h  = (_Float16*)ws;  ws += (size_t)M * C * sizeof(_Float16);
  float*    x1      = (float*)ws;     ws += (size_t)M * C * sizeof(float);
  _Float16* hid_h   = qkv_h;  // reuse qkv(24MB)+attn(8MB) contiguous span for [M][HID]

  auto cast = [&](const float* s, _Float16* d, size_t n) {
    cast_f32_f16_kernel<<<dim3((unsigned)((n / 4 + 255) / 256)), 256, 0, stream>>>(s, d, (int)(n / 4));
  };
  cast(w_qkv,  wqkv_h,  (size_t)3 * C * C);
  cast(w_proj, wproj_h, (size_t)C * C);
  cast(w_fc1,  wfc1_h,  (size_t)HID * C);
  cast(w_fc2,  wfc2_h,  (size_t)C * HID);

  ln_f16_kernel<<<M, 256, 0, stream>>>(x, ln1_g, ln1_b, ln_h, C);
  gemm_wmma_kernel<<<dim3(3 * C / TN, M / TM), 256, 0, stream>>>(
      ln_h, wqkv_h, qkv_h, nullptr, nullptr, nullptr, M, 3 * C, C, 0);
  attn_flash_kernel<<<dim3(B * H, N / 128), 256, 0, stream>>>(qkv_h, attn_h, N, C, H);
  gemm_wmma_kernel<<<dim3(C / TN, M / TM), 256, 0, stream>>>(
      attn_h, wproj_h, nullptr, x1, b_proj, x, M, C, C, 2);
  ln_f16_kernel<<<M, 256, 0, stream>>>(x1, ln2_g, ln2_b, ln_h, C);
  gemm_wmma_kernel<<<dim3(HID / TN, M / TM), 256, 0, stream>>>(
      ln_h, wfc1_h, hid_h, nullptr, b_fc1, nullptr, M, HID, C, 1);
  gemm_wmma_kernel<<<dim3(C / TN, M / TM), 256, 0, stream>>>(
      hid_h, wfc2_h, nullptr, out, b_fc2, x1, M, C, HID, 2);
}
